// TopK_9912784519967
// MI455X (gfx1250) — matrix operand order repaired and checked by reading, not compiled, not run on hardware
//
#include <hip/hip_runtime.h>
#include <hip/hip_bf16.h>
#include <math.h>

// ---------------- problem constants ----------------
#define BG   64      // graphs
#define NN   1024    // nodes per graph
#define KK   820     // kept nodes per graph
#define HH   128     // feature/hidden width
#define CC   10      // classes

typedef __attribute__((ext_vector_type(16))) _Float16 v16h;
typedef __attribute__((ext_vector_type(8)))  _Float16 v8h;
typedef __attribute__((ext_vector_type(8)))  float    v8f;

// ---------------- small utility kernels ----------------
__global__ void fill_kernel(float* __restrict__ p, float v, int n) {
    int i = blockIdx.x * blockDim.x + threadIdx.x;
    if (i < n) p[i] = v;
}

// f32 -> f16 conversion, optionally scaling row m by 1/max(deg[m],1) (fused mean-normalize)
__global__ void cvt_kernel(const float* __restrict__ in, _Float16* __restrict__ out,
                           const float* __restrict__ deg, int K, int total) {
    int i = blockIdx.x * blockDim.x + threadIdx.x;
    if (i >= total) return;
    float v = in[i];
    if (deg) v *= 1.0f / fmaxf(deg[i / K], 1.0f);
    out[i] = (_Float16)v;
}

// per-edge degree accumulation (emask == 1 for layers 1&2)
__global__ void deg_kernel(const int* __restrict__ dst, float* __restrict__ deg, int E) {
    int i = blockIdx.x * blockDim.x + threadIdx.x;
    if (i < E) atomicAdd(&deg[dst[i]], 1.0f);
}

// degree for pooled graph: edges valid only if both endpoints kept
__global__ void deg3_kernel(const int* __restrict__ src, const int* __restrict__ dst,
                            const int* __restrict__ mapping, float* __restrict__ deg, int E) {
    int i = blockIdx.x * blockDim.x + threadIdx.x;
    if (i < E) {
        int s = mapping[src[i]], d = mapping[dst[i]];
        if (s >= 0 && d >= 0) atomicAdd(&deg[d], 1.0f);
    }
}

// message scatter: agg[dst] += x[src]   (one block per edge, 128 feature lanes)
__global__ void scatter_kernel(const int* __restrict__ src, const int* __restrict__ dst,
                               const float* __restrict__ X, float* __restrict__ agg) {
    int e = blockIdx.x, f = threadIdx.x;
    int s = src[e], d = dst[e];
    atomicAdd(&agg[(size_t)d * HH + f], X[(size_t)s * HH + f]);
}

// filtered scatter for pooled graph
__global__ void scatter3_kernel(const int* __restrict__ src, const int* __restrict__ dst,
                                const int* __restrict__ mapping,
                                const float* __restrict__ X, float* __restrict__ agg) {
    int e = blockIdx.x, f = threadIdx.x;
    int s = mapping[src[e]], d = mapping[dst[e]];
    if (s < 0 || d < 0) return;
    atomicAdd(&agg[(size_t)d * HH + f], X[(size_t)s * HH + f]);
}

// global mean pool into the [64, 384] JK-concat buffer
__global__ void mean_pool_kernel(const float* __restrict__ X, float* __restrict__ xs_cat,
                                 int rows, int colOff, float inv) {
    int g = blockIdx.x, f = threadIdx.x;          // 64 blocks x 128 threads
    const float* p = X + (size_t)g * rows * HH + f;
    float s = 0.0f;
    for (int r = 0; r < rows; ++r) s += p[(size_t)r * HH];
    xs_cat[g * (3 * HH) + colOff + f] = s * inv;
}

// gather kept nodes scaled by score: xp[j] = x2[old[j]] * vals[j]
__global__ void gather_kernel(const float* __restrict__ x2, const int* __restrict__ oldidx,
                              const float* __restrict__ vals, float* __restrict__ xp) {
    int n = blockIdx.x, f = threadIdx.x;
    xp[(size_t)n * HH + f] = x2[(size_t)oldidx[n] * HH + f] * vals[n];
}

// ---------------- TopK pooling: score + per-graph bitonic sort ----------------
__global__ void __launch_bounds__(1024)
topk_kernel(const float* __restrict__ x2, const float* __restrict__ p_w,
            int* __restrict__ oldidx, float* __restrict__ vals, int* __restrict__ mapping) {
    __shared__ float pw[HH];
    __shared__ float sc[NN];
    __shared__ int   id[NN];
    __shared__ float ninv;
    int g = blockIdx.x, t = threadIdx.x;
    if (t < HH) pw[t] = p_w[t];
    __syncthreads();
    if (t == 0) {
        float s = 0.0f;
        for (int i = 0; i < HH; ++i) s += pw[i] * pw[i];
        ninv = 1.0f / sqrtf(s);
    }
    __syncthreads();
    // score = tanh(x . w / ||w||)
    const float* xr = x2 + ((size_t)g * NN + t) * HH;
    float s = 0.0f;
    for (int i = 0; i < HH; ++i) s += xr[i] * pw[i];
    sc[t] = tanhf(s * ninv);
    id[t] = t;
    __syncthreads();
    // bitonic sort, descending (order within top-K is irrelevant: head mean-pools)
    for (int k = 2; k <= NN; k <<= 1) {
        for (int j = k >> 1; j > 0; j >>= 1) {
            int ixj = t ^ j;
            if (ixj > t) {
                bool desc = ((t & k) == 0);
                float a = sc[t], b = sc[ixj];
                bool sw = desc ? (a < b) : (a > b);
                if (sw) {
                    sc[t] = b; sc[ixj] = a;
                    int tmp = id[t]; id[t] = id[ixj]; id[ixj] = tmp;
                }
            }
            __syncthreads();
        }
    }
    mapping[g * NN + t] = -1;
    __syncthreads();
    if (t < KK) {
        int o = g * NN + id[t];
        oldidx[g * KK + t] = o;
        vals[g * KK + t]   = sc[t];
        mapping[o]         = g * KK + t;
    }
}

// ---------------- fused WMMA GEMM: one wave per 16x128 output strip ----------------
// Y[m,n] = act( sum_k A1[m,k]*W1[n,k] (+ sum_k A2[m,k]*W2[n,k]) + bias[n] )
// A inputs pre-converted to f16 (deg-normalization fused into the conversion pass).
// KD (K depth) and DUAL are compile-time: the kb loop fully unrolls, every load
// and store becomes base-pointer + immediate-offset, and the WMMA stream is
// branch-free (64 WMMAs for <128,true>, 96 for <384,false>).
//
// Fragment layouts used by this kernel (wave32, contiguous k = 16*hi + s, same for A and B):
//  A 16x32 f16: lane<16 row r holds K kb+0..15; lane>=16 holds
//               kb+16..31  -> two 16B chunks per fragment.
//  B 32x16 f16: lane<16 col n holds K kb+0..15 (one contiguous W row chunk);
//               lane>=16 holds kb+16..31 -> one 32B-aligned v16h load.
//  C/D f32:     VGPR i = row m0 + (hi?8:0) + i, column = n0 + r.
template <int KD, bool DUAL>
__global__ void __launch_bounds__(256)
gemm_wmma_kernel(const _Float16* __restrict__ A1, const _Float16* __restrict__ W1,
                 const _Float16* __restrict__ A2, const _Float16* __restrict__ W2,
                 const float* __restrict__ bias, float* __restrict__ Y,
                 int M, int doRelu) {
    int wave  = threadIdx.x >> 5;
    int lane  = threadIdx.x & 31;
    int strip = blockIdx.x * 8 + wave;
    if (strip >= (M >> 4)) return;              // wave-uniform exit: EXEC stays all-ones
    int m0 = strip << 4;
    int r  = lane & 15;
    int hi = lane >> 4;

    const _Float16* pA1 = A1 + (size_t)(m0 + r) * KD + (hi ? 16 : 0);
    const _Float16* pA2 = DUAL ? A2 + (size_t)(m0 + r) * KD + (hi ? 16 : 0) : nullptr;
    const _Float16* pB1 = W1 + (size_t)r * KD + (hi ? 16 : 0);
    const _Float16* pB2 = DUAL ? W2 + (size_t)r * KD + (hi ? 16 : 0) : nullptr;

    v8f acc[8];
#pragma unroll
    for (int nt = 0; nt < 8; ++nt)
        acc[nt] = (v8f){0.f, 0.f, 0.f, 0.f, 0.f, 0.f, 0.f, 0.f};

#pragma unroll
    for (int kb = 0; kb < KD; kb += 32) {
        union { v16h v; v8h h[2]; } a1u, a2u;
        a1u.h[0] = *(const v8h*)(pA1 + kb);
        a1u.h[1] = *(const v8h*)(pA1 + kb + 8);
        v16h a1 = a1u.v;
        v16h a2;
        if (DUAL) {
            a2u.h[0] = *(const v8h*)(pA2 + kb);
            a2u.h[1] = *(const v8h*)(pA2 + kb + 8);
            a2 = a2u.v;
        }
#pragma unroll
        for (int nt = 0; nt < 8; ++nt) {
            v16h b1 = *(const v16h*)(pB1 + kb + nt * (16 * KD));
            acc[nt] = __builtin_amdgcn_wmma_f32_16x16x32_f16(false, a1, false, b1,
                                                             (short)0, acc[nt], false, false);
            if (DUAL) {
                v16h b2 = *(const v16h*)(pB2 + kb + nt * (16 * KD));
                acc[nt] = __builtin_amdgcn_wmma_f32_16x16x32_f16(false, a2, false, b2,
                                                                 (short)0, acc[nt], false, false);
            }
        }
    }

    int rbase = m0 + (hi ? 8 : 0);
    float* pY = Y + (size_t)rbase * HH + r;
#pragma unroll
    for (int nt = 0; nt < 8; ++nt) {
        float bv = bias[nt * 16 + r];
#pragma unroll
        for (int i = 0; i < 8; ++i) {
            float v = acc[nt][i] + bv;
            if (doRelu) v = fmaxf(v, 0.0f);
            pY[(size_t)i * HH + nt * 16] = v;
        }
    }
}

// ---------------- classifier head: logits + log_softmax ----------------
__global__ void head_kernel(const float* __restrict__ h, const float* __restrict__ W2,
                            const float* __restrict__ b2, float* __restrict__ out) {
    __shared__ float logit[CC];
    int g = blockIdx.x, t = threadIdx.x;  // 32 threads
    if (t < CC) {
        const float* hr = h + (size_t)g * HH;
        const float* w  = W2 + (size_t)t * HH;
        float s = b2[t];
        for (int i = 0; i < HH; ++i) s += hr[i] * w[i];
        logit[t] = s;
    }
    __syncthreads();
    if (t == 0) {
        float mx = logit[0];
        for (int c = 1; c < CC; ++c) mx = fmaxf(mx, logit[c]);
        float se = 0.0f;
        for (int c = 0; c < CC; ++c) se += expf(logit[c] - mx);
        float lse = mx + logf(se);
        for (int c = 0; c < CC; ++c) out[g * CC + c] = logit[c] - lse;
    }
}

// ---------------- orchestration ----------------
extern "C" void kernel_launch(void* const* d_in, const int* in_sizes, int n_in,
                              void* d_out, int out_size, void* d_ws, size_t ws_size,
                              hipStream_t stream) {
    const float* x       = (const float*)d_in[0];
    const int*   eidx    = (const int*)d_in[1];
    const float* W_rel1  = (const float*)d_in[3];
    const float* b_rel1  = (const float*)d_in[4];
    const float* W_root1 = (const float*)d_in[5];
    const float* W_rel2  = (const float*)d_in[6];
    const float* b_rel2  = (const float*)d_in[7];
    const float* W_root2 = (const float*)d_in[8];
    const float* W_rel3  = (const float*)d_in[9];
    const float* b_rel3  = (const float*)d_in[10];
    const float* W_root3 = (const float*)d_in[11];
    const float* p_w     = (const float*)d_in[12];
    const float* W_l1    = (const float*)d_in[13];
    const float* b_l1    = (const float*)d_in[14];
    const float* W_l2    = (const float*)d_in[15];
    const float* b_l2    = (const float*)d_in[16];
    float* out = (float*)d_out;

    const int E  = in_sizes[1] / 2;
    const int M1 = BG * NN;   // 65536
    const int M2 = BG * KK;   // 52480
    const int* src = eidx;
    const int* dst = eidx + E;

    // workspace carve-out (256B aligned)
    char*  ws  = (char*)d_ws;
    size_t off = 0;
    auto alloc = [&](size_t bytes) -> void* {
        void* p = ws + off;
        off = (off + bytes + 255) & ~(size_t)255;
        return p;
    };
    float*     agg     = (float*)    alloc((size_t)M1 * HH * 4);  // shared message buffer
    float*     buf1    = (float*)    alloc((size_t)M1 * HH * 4);  // x1, reused as xp
    float*     buf2    = (float*)    alloc((size_t)M1 * HH * 4);  // x2, reused as x3
    _Float16*  agg_h   = (_Float16*) alloc((size_t)M1 * HH * 2);  // f16 agg (deg-normalized)
    _Float16*  feat_h  = (_Float16*) alloc((size_t)M1 * HH * 2);  // f16 root-side features
    float*     deg     = (float*)    alloc((size_t)M1 * 4);
    float*     deg2    = (float*)    alloc((size_t)M2 * 4);
    int*       oldidx  = (int*)      alloc((size_t)M2 * 4);
    float*     vals    = (float*)    alloc((size_t)M2 * 4);
    int*       mapping = (int*)      alloc((size_t)M1 * 4);
    float*     xs_cat  = (float*)    alloc((size_t)BG * 3 * HH * 4);
    _Float16*  xs_h    = (_Float16*) alloc((size_t)BG * 3 * HH * 2);
    float*     hbuf    = (float*)    alloc((size_t)BG * HH * 4);
    _Float16*  wr1h    = (_Float16*) alloc((size_t)HH * HH * 2);
    _Float16*  wo1h    = (_Float16*) alloc((size_t)HH * HH * 2);
    _Float16*  wr2h    = (_Float16*) alloc((size_t)HH * HH * 2);
    _Float16*  wo2h    = (_Float16*) alloc((size_t)HH * HH * 2);
    _Float16*  wr3h    = (_Float16*) alloc((size_t)HH * HH * 2);
    _Float16*  wo3h    = (_Float16*) alloc((size_t)HH * HH * 2);
    _Float16*  wl1h    = (_Float16*) alloc((size_t)HH * 3 * HH * 2);

    const int T = 256;
    const int nW  = HH * HH;            // 16384
    const int nA1 = M1 * HH, nA2 = M2 * HH;
    auto blocks = [](int n, int t) { return (n + t - 1) / t; };
    const int gemmB1 = ((M1 >> 4) + 7) / 8;
    const int gemmB2 = ((M2 >> 4) + 7) / 8;

    // ---- weight conversions (f32 -> f16, done every launch; deterministic) ----
    cvt_kernel<<<blocks(nW, T), T, 0, stream>>>(W_rel1,  wr1h, nullptr, HH, nW);
    cvt_kernel<<<blocks(nW, T), T, 0, stream>>>(W_root1, wo1h, nullptr, HH, nW);
    cvt_kernel<<<blocks(nW, T), T, 0, stream>>>(W_rel2,  wr2h, nullptr, HH, nW);
    cvt_kernel<<<blocks(nW, T), T, 0, stream>>>(W_root2, wo2h, nullptr, HH, nW);
    cvt_kernel<<<blocks(nW, T), T, 0, stream>>>(W_rel3,  wr3h, nullptr, HH, nW);
    cvt_kernel<<<blocks(nW, T), T, 0, stream>>>(W_root3, wo3h, nullptr, HH, nW);
    cvt_kernel<<<blocks(3 * nW, T), T, 0, stream>>>(W_l1, wl1h, nullptr, 3 * HH, 3 * nW);

    // ---- layer 1 ----
    fill_kernel<<<blocks(nA1, T), T, 0, stream>>>(agg, 0.0f, nA1);
    fill_kernel<<<blocks(M1, T), T, 0, stream>>>(deg, 0.0f, M1);
    deg_kernel<<<blocks(E, T), T, 0, stream>>>(dst, deg, E);
    scatter_kernel<<<E, HH, 0, stream>>>(src, dst, x, agg);
    cvt_kernel<<<blocks(nA1, T), T, 0, stream>>>(agg, agg_h, deg, HH, nA1);
    cvt_kernel<<<blocks(nA1, T), T, 0, stream>>>(x, feat_h, nullptr, HH, nA1);
    gemm_wmma_kernel<HH, true><<<gemmB1, 256, 0, stream>>>(agg_h, wr1h, feat_h, wo1h,
                                                           b_rel1, buf1, M1, 1);
    mean_pool_kernel<<<BG, HH, 0, stream>>>(buf1, xs_cat, NN, 0, 1.0f / (float)NN);

    // ---- layer 2 ----
    fill_kernel<<<blocks(nA1, T), T, 0, stream>>>(agg, 0.0f, nA1);
    scatter_kernel<<<E, HH, 0, stream>>>(src, dst, buf1, agg);
    cvt_kernel<<<blocks(nA1, T), T, 0, stream>>>(agg, agg_h, deg, HH, nA1);
    cvt_kernel<<<blocks(nA1, T), T, 0, stream>>>(buf1, feat_h, nullptr, HH, nA1);
    gemm_wmma_kernel<HH, true><<<gemmB1, 256, 0, stream>>>(agg_h, wr2h, feat_h, wo2h,
                                                           b_rel2, buf2, M1, 1);
    mean_pool_kernel<<<BG, HH, 0, stream>>>(buf2, xs_cat, NN, HH, 1.0f / (float)NN);

    // ---- TopK pooling ----
    topk_kernel<<<BG, NN, 0, stream>>>(buf2, p_w, oldidx, vals, mapping);
    gather_kernel<<<M2, HH, 0, stream>>>(buf2, oldidx, vals, buf1 /* xp */);

    // ---- layer 3 on pooled graph ----
    fill_kernel<<<blocks(nA2, T), T, 0, stream>>>(agg, 0.0f, nA2);
    fill_kernel<<<blocks(M2, T), T, 0, stream>>>(deg2, 0.0f, M2);
    deg3_kernel<<<blocks(E, T), T, 0, stream>>>(src, dst, mapping, deg2, E);
    scatter3_kernel<<<E, HH, 0, stream>>>(src, dst, mapping, buf1, agg);
    cvt_kernel<<<blocks(nA2, T), T, 0, stream>>>(agg, agg_h, deg2, HH, nA2);
    cvt_kernel<<<blocks(nA2, T), T, 0, stream>>>(buf1, feat_h, nullptr, HH, nA2);
    gemm_wmma_kernel<HH, true><<<gemmB2, 256, 0, stream>>>(agg_h, wr3h, feat_h, wo3h,
                                                           b_rel3, buf2 /* x3 */, M2, 1);
    mean_pool_kernel<<<BG, HH, 0, stream>>>(buf2, xs_cat, KK, 2 * HH, 1.0f / (float)KK);

    // ---- head: lin1 (K=384, single GEMM) + lin2/log_softmax ----
    cvt_kernel<<<blocks(BG * 3 * HH, T), T, 0, stream>>>(xs_cat, xs_h, nullptr,
                                                         3 * HH, BG * 3 * HH);
    gemm_wmma_kernel<3 * HH, false><<<1, 256, 0, stream>>>(xs_h, wl1h, nullptr, nullptr,
                                                           b_l1, hbuf, BG, 1);
    head_kernel<<<BG, 32, 0, stream>>>(hbuf, W_l2, b_l2, out);

    (void)n_in; (void)out_size; (void)ws_size;
}
